// GAT_42545946034486
// MI455X (gfx1250) — compile-verified, matrix operand order checked by reading, not run on hardware
//
#include <hip/hip_runtime.h>
#include <hip/hip_bf16.h>
#include <stdint.h>

typedef __attribute__((ext_vector_type(16))) _Float16 v16h;
typedef __attribute__((ext_vector_type(4)))  _Float16 v4h;
typedef __attribute__((ext_vector_type(4)))  float    v4f;
typedef __attribute__((ext_vector_type(8)))  float    v8f;

#define NNODES   50000
#define NEDGES   800000
#define NDF      256
#define HEADS    4
#define CH1      64
#define CH2      128
#define NEG_SLOPE 0.2f

// ---------------- order-preserving float<->uint map for atomicMax ----------------
__device__ __forceinline__ unsigned fmax_map(float f) {
    unsigned u = __float_as_uint(f);
    return (u & 0x80000000u) ? ~u : (u | 0x80000000u);
}
__device__ __forceinline__ float fmax_unmap(unsigned u) {
    return (u & 0x80000000u) ? __uint_as_float(u & 0x7FFFFFFFu) : __uint_as_float(~u);
}

// ---------------- utility kernels ----------------
__global__ void k_zero_f32(float* __restrict__ p, long n) {
    long i = (long)blockIdx.x * blockDim.x + threadIdx.x;
    if (i < n) p[i] = 0.0f;
}
__global__ void k_init_amax(unsigned* __restrict__ p, int n) {
    int i = blockIdx.x * blockDim.x + threadIdx.x;
    if (i < n) p[i] = 0x007FFFFFu;   // fmax_map(-inf)
}

// ---------------- WMMA GEMM: C[M,N] = A[M,K] @ B[K,N] (+bias) ----------------
// 128 threads = 4 waves. Block tile 128x64. Wave w owns rows [32w, 32w+32) as two
// 16-row fragments; B fragment loaded once per col-tile, used by both -> 8 WMMA/K-step.
#define TM 128
#define TN 64
#define TK 32
#define LSTR 40   // halves; 80B row stride keeps every fragment chunk 16B-aligned

__global__ __launch_bounds__(128)
void k_gemm_wmma(const float* __restrict__ A, const float* __restrict__ B,
                 const float* __restrict__ bias, float* __restrict__ C,
                 int M, int N, int K, int apply_bias)
{
    __shared__ __align__(16) _Float16 sA[TM * LSTR];   // [row][k]
    __shared__ __align__(16) _Float16 sB[TN * LSTR];   // [col][k] (transposed)

    const int t    = threadIdx.x;
    const int wave = t >> 5;
    const int lane = t & 31;
    const int lrow = lane & 15;
    const int kb   = (lane >= 16) ? 8 : 0;     // K split per ISA 16-bit frag layout
    const int m0   = blockIdx.x * TM;
    const int n0   = blockIdx.y * TN;

    // ---- per-thread staging maps (computed once; 64-bit math hoisted out of K loop)
    // A tile: 128x32 floats = 1024 float4 chunks; chunk c: row=c>>3, k4=(c&7)*4
    const float* aP[8]; int aL[8];
    #pragma unroll
    for (int i = 0; i < 8; ++i) {
        int c  = i * 128 + t;
        int r  = c >> 3;
        int k4 = (c & 7) * 4;
        int gr = m0 + r; if (gr > M - 1) gr = M - 1;   // clamp: branch-free staging
        aP[i] = A + (size_t)gr * K + k4;
        aL[i] = r * LSTR + k4;
    }
    // B tile: 32x64 floats = 512 float4 chunks; chunk c: k=c>>4, n4=(c&15)*4
    const float* bP[4]; int bK[4], bN4[4];
    #pragma unroll
    for (int i = 0; i < 4; ++i) {
        int c  = i * 128 + t;
        bK[i]  = c >> 4;
        bN4[i] = (c & 15) * 4;
        bP[i]  = B + (size_t)bK[i] * N + n0 + bN4[i];
    }

    v8f acc[8];   // [0..3]: rows 32w+0..15 x 4 col tiles ; [4..7]: rows 32w+16..31
    #pragma unroll
    for (int i = 0; i < 8; ++i)
        #pragma unroll
        for (int j = 0; j < 8; ++j) acc[i][j] = 0.0f;

    for (int k0 = 0; k0 < K; k0 += TK) {
        // stage A (vector loads, packed 8B LDS stores)
        #pragma unroll
        for (int i = 0; i < 8; ++i) {
            v4f v = *(const v4f*)(aP[i] + k0);
            v4h hv = __builtin_convertvector(v, v4h);
            *(v4h*)&sA[aL[i]] = hv;
        }
        // stage B transposed (vector load, 4x b16 scatter)
        #pragma unroll
        for (int i = 0; i < 4; ++i) {
            v4f v = *(const v4f*)(bP[i] + (size_t)k0 * N);
            v4h hv = __builtin_convertvector(v, v4h);
            #pragma unroll
            for (int j = 0; j < 4; ++j)
                sB[(bN4[i] + j) * LSTR + bK[i]] = hv[j];
        }
        __syncthreads();

        // two A fragments for this wave
        v16h af0, af1;
        {
            const _Float16* pa = &sA[(wave * 32 + lrow) * LSTR + kb];
            *((uint4*)&af0)     = *(const uint4*)pa;
            *((uint4*)&af0 + 1) = *(const uint4*)(pa + 16);
            pa += 16 * LSTR;
            *((uint4*)&af1)     = *(const uint4*)pa;
            *((uint4*)&af1 + 1) = *(const uint4*)(pa + 16);
        }
        #pragma unroll
        for (int tn = 0; tn < 4; ++tn) {
            v16h bf;
            const _Float16* pb = &sB[(tn * 16 + lrow) * LSTR + kb];
            *((uint4*)&bf)     = *(const uint4*)pb;
            *((uint4*)&bf + 1) = *(const uint4*)(pb + 16);
            acc[tn]     = __builtin_amdgcn_wmma_f32_16x16x32_f16(
                              false, af0, false, bf, (short)0, acc[tn],     false, false);
            acc[tn + 4] = __builtin_amdgcn_wmma_f32_16x16x32_f16(
                              false, af1, false, bf, (short)0, acc[tn + 4], false, false);
        }
        __syncthreads();
    }

    // ---- store D: VGPR r holds M = r + 8*(lane>=16), N = lane&15
    const int rb0 = m0 + wave * 32 + ((lane >= 16) ? 8 : 0);
    if (m0 + TM <= M) {           // full tile: no per-store bounds checks
        #pragma unroll
        for (int half = 0; half < 2; ++half) {
            int rbase = rb0 + half * 16;
            #pragma unroll
            for (int tn = 0; tn < 4; ++tn) {
                int col = n0 + tn * 16 + lrow;
                float bv = apply_bias ? bias[col] : 0.0f;
                #pragma unroll
                for (int r = 0; r < 8; ++r)
                    C[(size_t)(rbase + r) * N + col] = acc[half * 4 + tn][r] + bv;
            }
        }
    } else {
        #pragma unroll
        for (int half = 0; half < 2; ++half) {
            int rbase = rb0 + half * 16;
            #pragma unroll
            for (int tn = 0; tn < 4; ++tn) {
                int col = n0 + tn * 16 + lrow;
                float bv = apply_bias ? bias[col] : 0.0f;
                #pragma unroll
                for (int r = 0; r < 8; ++r) {
                    int row = rbase + r;
                    if (row < M) C[(size_t)row * N + col] = acc[half * 4 + tn][r] + bv;
                }
            }
        }
    }
}

// ---------------- per-node attention logits: out[n,h] = sum_c feat[n,h,c]*a[h,c] ----------------
__global__ void k_node_logit(const float* __restrict__ feat, const float* __restrict__ avec,
                             float* __restrict__ out, int n, int Hn, int Cn)
{
    int i = blockIdx.x * blockDim.x + threadIdx.x;
    if (i >= n * Hn) return;
    int node = i / Hn, h = i % Hn;
    const float* f = feat + ((size_t)node * Hn + h) * Cn;
    const float* a = avec + (size_t)h * Cn;
    float s = 0.0f;
    for (int c = 0; c < Cn; c += 4) {
        v4f fv = *(const v4f*)(f + c);
        v4f av = *(const v4f*)(a + c);
        s += fv[0]*av[0] + fv[1]*av[1] + fv[2]*av[2] + fv[3]*av[3];
    }
    out[i] = s;
}

// ---------------- edge pass 1: alpha = leakyrelu(as[src]+ad[dst]); segment max ----------------
__global__ void k_edge_alpha(const float* __restrict__ as, const float* __restrict__ ad,
                             const int* __restrict__ src, const int* __restrict__ dst,
                             float* __restrict__ alpha, unsigned* __restrict__ amax,
                             int E, int Hn)
{
    long i = (long)blockIdx.x * blockDim.x + threadIdx.x;
    if (i >= (long)E * Hn) return;
    int e = (int)(i / Hn), h = (int)(i % Hn);
    int s = src[e], d = dst[e];
    float a = as[s * Hn + h] + ad[d * Hn + h];
    a = (a > 0.0f) ? a : NEG_SLOPE * a;
    alpha[i] = a;
    atomicMax(&amax[d * Hn + h], fmax_map(a));
}

// ---------------- edge pass 2: ealpha = exp(alpha - amax[dst]); segment sum ----------------
__global__ void k_edge_exp(float* __restrict__ alpha, const unsigned* __restrict__ amax,
                           const int* __restrict__ dst, float* __restrict__ denom,
                           int E, int Hn)
{
    long i = (long)blockIdx.x * blockDim.x + threadIdx.x;
    if (i >= (long)E * Hn) return;
    int e = (int)(i / Hn), h = (int)(i % Hn);
    int d = dst[e];
    float m  = fmax_unmap(amax[d * Hn + h]);
    float ev = __expf(alpha[i] - m);
    alpha[i] = ev;                                // overwrite in place
    atomicAdd(&denom[d * Hn + h], ev);
}

// ---------------- edge pass 3: agg[dst] += hs[src] * attn ----------------
__global__ void k_edge_scatter(const float* __restrict__ ealpha, const float* __restrict__ denom,
                               const int* __restrict__ src, const int* __restrict__ dst,
                               const float* __restrict__ hs, float* __restrict__ agg,
                               int E, int Hn, int Cn)
{
    long i = (long)blockIdx.x * blockDim.x + threadIdx.x;
    if (i >= (long)E * Hn) return;
    int e = (int)(i / Hn), h = (int)(i % Hn);
    int sn = src[e], dn = dst[e];
    float attn = ealpha[i] / (denom[dn * Hn + h] + 1e-16f);
    const float* sp = hs  + ((size_t)sn * Hn + h) * Cn;
    float*       dp = agg + ((size_t)dn * Hn + h) * Cn;
    for (int c = 0; c < Cn; c += 4) {
        v4f v = *(const v4f*)(sp + c);
        atomicAdd(dp + c + 0, v[0] * attn);
        atomicAdd(dp + c + 1, v[1] * attn);
        atomicAdd(dp + c + 2, v[2] * attn);
        atomicAdd(dp + c + 3, v[3] * attn);
    }
}

// ---------------- h = relu(agg1 + b1 + lin1 + lin1_b)  (F = H*C1 = 256) ----------------
__global__ void k_combine1(const float* __restrict__ agg, const float* __restrict__ lin,
                           const float* __restrict__ b1, const float* __restrict__ lin_b,
                           float* __restrict__ h, long n, int F)
{
    long i = (long)blockIdx.x * blockDim.x + threadIdx.x;
    if (i >= n) return;
    int f = (int)(i % F);
    float v = agg[i] + lin[i] + b1[f] + lin_b[f];
    h[i] = (v > 0.0f) ? v : 0.0f;
}

// ---------------- hfinal[n,c] = mean_h(agg2[n,h,c]) + b2[c] + lin2[n,c] + lin2_b[c] ----------------
__global__ void k_combine2(const float* __restrict__ agg2, const float* __restrict__ lin2,
                           const float* __restrict__ b2, const float* __restrict__ lin2_b,
                           float* __restrict__ out, int n, int Hn, int Cn)
{
    long i = (long)blockIdx.x * blockDim.x + threadIdx.x;
    if (i >= (long)n * Cn) return;
    int node = (int)(i / Cn), c = (int)(i % Cn);
    const float* a = agg2 + (size_t)node * Hn * Cn + c;
    float s = 0.0f;
    #pragma unroll
    for (int h = 0; h < HEADS; ++h) s += a[(size_t)h * Cn];
    out[i] = s * (1.0f / Hn) + b2[c] + lin2[i] + lin2_b[c];
}

// =====================================================================================
extern "C" void kernel_launch(void* const* d_in, const int* in_sizes, int n_in,
                              void* d_out, int out_size, void* d_ws, size_t ws_size,
                              hipStream_t stream)
{
    const float* x     = (const float*)d_in[0];
    const int*   eidx  = (const int*)  d_in[1];
    const float* W1s   = (const float*)d_in[2];
    const float* W1d   = (const float*)d_in[3];
    const float* a1s   = (const float*)d_in[4];
    const float* a1d   = (const float*)d_in[5];
    const float* b1    = (const float*)d_in[6];
    const float* lin1W = (const float*)d_in[7];
    const float* lin1b = (const float*)d_in[8];
    const float* W2    = (const float*)d_in[9];
    const float* a2s   = (const float*)d_in[10];
    const float* a2d   = (const float*)d_in[11];
    const float* b2    = (const float*)d_in[12];
    const float* lin2W = (const float*)d_in[13];
    const float* lin2b = (const float*)d_in[14];
    const float* outW  = (const float*)d_in[15];
    const float* outb  = (const float*)d_in[16];

    const int* src = eidx;
    const int* dst = eidx + NEDGES;

    const long NH  = (long)NNODES * HEADS;           //   200,000
    const long NF1 = (long)NNODES * HEADS * CH1;     // 12,800,000
    const long NF2 = (long)NNODES * HEADS * CH2;     // 25,600,000
    const long NC2 = (long)NNODES * CH2;             //  6,400,000
    const long EH  = (long)NEDGES * HEADS;           //  3,200,000

    float* ws = (float*)d_ws;
    // layer-1 region (reused by layer 2 afterwards)
    float* hs1   = ws + 0;            // NF1
    float* hd1   = ws + NF1;          // NF1
    float* lin1o = ws + 2 * NF1;      // NF1
    float* agg1  = ws + 3 * NF1;      // NF1
    float* h     = ws + 4 * NF1;      // NF1  (persists)
    // layer-2 reuse: h2 over {hs1,hd1}, agg2 over {lin1o,agg1}
    float* h2    = ws + 0;            // NF2 == 2*NF1
    float* agg2  = ws + 2 * NF1;      // NF2
    float* lin2o = ws + 5 * NF1;                     // NC2
    float* hfin  = ws + 5 * NF1 + NC2;               // NC2
    // small buffers
    float*    asb   = ws + 5 * NF1 + 2 * NC2;        // NH
    float*    adb   = asb + NH;                      // NH
    unsigned* amaxu = (unsigned*)(adb + NH);         // NH
    float*    denom = (float*)(amaxu + NH);          // NH
    float*    alpha = denom + NH;                    // EH

    const dim3 blk128(128), blk256(256);
    const int gNH  = (int)((NH  + 255) / 256);
    const int gEH  = (int)((EH  + 255) / 256);

    // ---------- Layer 1 GEMMs: [50000,256] x [256,256] ----------
    {
        dim3 g((NNODES + TM - 1) / TM, NDF / TN);
        k_gemm_wmma<<<g, blk128, 0, stream>>>(x, W1s,   nullptr, hs1,   NNODES, NDF, NDF, 0);
        k_gemm_wmma<<<g, blk128, 0, stream>>>(x, W1d,   nullptr, hd1,   NNODES, NDF, NDF, 0);
        k_gemm_wmma<<<g, blk128, 0, stream>>>(x, lin1W, nullptr, lin1o, NNODES, NDF, NDF, 0);
    }
    // attention logits
    k_node_logit<<<gNH, blk256, 0, stream>>>(hs1, a1s, asb, NNODES, HEADS, CH1);
    k_node_logit<<<gNH, blk256, 0, stream>>>(hd1, a1d, adb, NNODES, HEADS, CH1);
    // init segment state + agg
    k_init_amax<<<gNH, blk256, 0, stream>>>(amaxu, (int)NH);
    k_zero_f32 <<<gNH, blk256, 0, stream>>>(denom, NH);
    k_zero_f32 <<<(int)((NF1 + 255) / 256), blk256, 0, stream>>>(agg1, NF1);
    // edge passes
    k_edge_alpha  <<<gEH, blk256, 0, stream>>>(asb, adb, src, dst, alpha, amaxu, NEDGES, HEADS);
    k_edge_exp    <<<gEH, blk256, 0, stream>>>(alpha, amaxu, dst, denom, NEDGES, HEADS);
    k_edge_scatter<<<gEH, blk256, 0, stream>>>(alpha, denom, src, dst, hs1, agg1, NEDGES, HEADS, CH1);
    // h = relu(agg1 + b1 + lin1 + lin1_b)
    k_combine1<<<(int)((NF1 + 255) / 256), blk256, 0, stream>>>(agg1, lin1o, b1, lin1b, h, NF1, HEADS * CH1);

    // ---------- Layer 2 GEMMs ----------
    {
        dim3 g2((NNODES + TM - 1) / TM, (HEADS * CH2) / TN);   // N = 512
        k_gemm_wmma<<<g2, blk128, 0, stream>>>(h, W2, nullptr, h2, NNODES, HEADS * CH2, HEADS * CH1, 0);
        dim3 gl((NNODES + TM - 1) / TM, CH2 / TN);             // N = 128
        k_gemm_wmma<<<gl, blk128, 0, stream>>>(h, lin2W, nullptr, lin2o, NNODES, CH2, HEADS * CH1, 0);
    }
    k_node_logit<<<gNH, blk256, 0, stream>>>(h2, a2s, asb, NNODES, HEADS, CH2);
    k_node_logit<<<gNH, blk256, 0, stream>>>(h2, a2d, adb, NNODES, HEADS, CH2);
    k_init_amax<<<gNH, blk256, 0, stream>>>(amaxu, (int)NH);
    k_zero_f32 <<<gNH, blk256, 0, stream>>>(denom, NH);
    k_zero_f32 <<<(int)((NF2 + 255) / 256), blk256, 0, stream>>>(agg2, NF2);
    k_edge_alpha  <<<gEH, blk256, 0, stream>>>(asb, adb, src, dst, alpha, amaxu, NEDGES, HEADS);
    k_edge_exp    <<<gEH, blk256, 0, stream>>>(alpha, amaxu, dst, denom, NEDGES, HEADS);
    k_edge_scatter<<<gEH, blk256, 0, stream>>>(alpha, denom, src, dst, h2, agg2, NEDGES, HEADS, CH2);
    // hfinal = mean_h(agg2) + b2 + lin2 + lin2_b
    k_combine2<<<(int)((NC2 + 255) / 256), blk256, 0, stream>>>(agg2, lin2o, b2, lin2b, hfin, NNODES, HEADS, CH2);

    // ---------- Output GEMM: [50000,128] x [128,64] + out_b ----------
    {
        dim3 go((NNODES + TM - 1) / TM, 64 / TN);
        k_gemm_wmma<<<go, blk128, 0, stream>>>(hfin, outW, outb, (float*)d_out, NNODES, 64, CH2, 1);
    }
}
